// WDGSNet_41807211659528
// MI455X (gfx1250) — matrix operand order repaired and checked by reading, not compiled, not run on hardware
//
#include <hip/hip_runtime.h>

// ---------------------------------------------------------------------------
// WDGSNet for MI455X (gfx1250): all GEMM-shaped ops via v_wmma_f32_16x16x32_bf16
// Register-blocked (A-fragment reuse x4) + vectorized b128 staging.
// ---------------------------------------------------------------------------
typedef __attribute__((ext_vector_type(16))) __bf16 v16bf;
typedef __attribute__((ext_vector_type(8)))  float  v8f;

union FragU  { v16bf v; unsigned int u[8]; };
union Pack4  { __bf16 h[4]; uint2 u2; };

// ---------------------------------------------------------------------------
// conv1x1 GEMM: Out[m, n] = sum_k X[m, k] * W[n, k] + bias[n]   (Nc == 64)
// Block = 256 threads = 8 waves; block tile 128(M) x 64(N);
// wave w owns rows [m0 + 16w, +16) and all 4 N-tiles (A-fragment reuse x4).
// ---------------------------------------------------------------------------
template <int K, bool ACCUM>
__global__ __launch_bounds__(256)
void conv_gemm_wmma(const float* __restrict__ X, int lda,
                    const float* __restrict__ W, const float* __restrict__ bias,
                    float* __restrict__ Out, int M)
{
    __shared__ __align__(16) __bf16 As[128 * 32];
    __shared__ __align__(16) __bf16 Bs[64 * 32];
    const int tid  = threadIdx.x;
    const int wave = tid >> 5;
    const int lane = tid & 31;
    const int m0   = blockIdx.x * 128;
    const int row  = lane & 15;
    const int hi   = lane >> 4;

    v8f acc[4] = {};
    for (int k0 = 0; k0 < K; k0 += 32) {
        __syncthreads();
        // stage A tile (128 x 32) as float4 chunks -> packed bf16 b64 stores
#pragma unroll
        for (int it = 0; it < 4; ++it) {
            int ch = tid + it * 256;          // 1024 chunks of 4
            int r  = ch >> 3, k4 = (ch & 7) << 2;
            int m  = m0 + r;
            float4 v = {};
            if (m < M) v = *(const float4*)&X[(size_t)m * lda + k0 + k4];
            Pack4 p;
            p.h[0] = (__bf16)v.x; p.h[1] = (__bf16)v.y;
            p.h[2] = (__bf16)v.z; p.h[3] = (__bf16)v.w;
            *(uint2*)&As[(r << 5) + k4] = p.u2;
        }
        // stage W tile (64 x 32) -- already in "B^T" (n-major, k-contig) form
#pragma unroll
        for (int it = 0; it < 2; ++it) {
            int ch = tid + it * 256;          // 512 chunks of 4
            int n  = ch >> 3, k4 = (ch & 7) << 2;
            float4 v = *(const float4*)&W[(size_t)n * K + k0 + k4];
            Pack4 p;
            p.h[0] = (__bf16)v.x; p.h[1] = (__bf16)v.y;
            p.h[2] = (__bf16)v.z; p.h[3] = (__bf16)v.w;
            *(uint2*)&Bs[(n << 5) + k4] = p.u2;
        }
        // prefetch next K slice into cache while we compute this one
        if (k0 + 32 < K) {
            int r = tid >> 1;
            int m = m0 + r; if (m >= M) m = M - 1;
            __builtin_prefetch(&X[(size_t)m * lda + k0 + 32 + ((tid & 1) << 4)], 0, 0);
            if (tid < 64) __builtin_prefetch(&W[(size_t)tid * K + k0 + 32], 0, 0);
        }
        __syncthreads();

        const __bf16* Ab = As + wave * 16 * 32;
        FragU a;
#pragma unroll
        for (int j = 0; j < 8; ++j) {
            int ka = ((j < 4) ? 0 : 16) + hi * 8 + ((j & 3) << 1); // A 16x32 bf16 layout
            a.u[j] = *(const unsigned int*)(Ab + (row << 5) + ka);
        }
#pragma unroll
        for (int nt = 0; nt < 4; ++nt) {
            const __bf16* Bb = Bs + nt * 16 * 32;
            FragU b;
#pragma unroll
            for (int j = 0; j < 8; ++j) {
                int kb = ((j < 4) ? 0 : 16) + hi * 8 + ((j & 3) << 1); // B 32x16 bf16 layout
                b.u[j] = *(const unsigned int*)(Bb + (row << 5) + kb);
            }
            acc[nt] = __builtin_amdgcn_wmma_f32_16x16x32_bf16(false, a.v, false, b.v,
                                                              (short)0, acc[nt], false, false);
        }
    }

#pragma unroll
    for (int nt = 0; nt < 4; ++nt) {
        const int n = nt * 16 + row;
#pragma unroll
        for (int r = 0; r < 8; ++r) {
            int m = m0 + wave * 16 + hi * 8 + r;  // C/D layout: M = r + 8*hi
            if (m < M) {
                size_t o = (size_t)m * 64 + n;
                float v = acc[nt][r] + bias[n];
                if (ACCUM) v += Out[o];
                Out[o] = v;
            }
        }
    }
}

// ---------------------------------------------------------------------------
// diffusion GEMM: Out_b[m, (t,c)] = sum_k Adj[k, m] * X_b[k, (t,c)]
// n = t*64 + c.  Strided addressing so results can land in concat buffers.
// Block = 256 threads = 8 waves; tile 16(M) x 512(N); wave w owns n-range
// [n0 + 64w, +64) as 4 tiles (A-fragment reuse x4).
// ---------------------------------------------------------------------------
template <int K>
__global__ __launch_bounds__(256)
void diffuse_gemm_wmma(const float* __restrict__ Adj, int ldadj, int M,
                       const float* __restrict__ X, int xBS, int xVS, int xTS,
                       float* __restrict__ Out, int oBS, int oVS, int oTS)
{
    __shared__ __align__(16) __bf16 As[16 * 32];
    __shared__ __align__(16) __bf16 Bs[512 * 32];
    const int tid  = threadIdx.x;
    const int wave = tid >> 5;
    const int lane = tid & 31;
    const int m0   = blockIdx.x * 16;
    const int n0   = blockIdx.y * 512;
    const int bb   = blockIdx.z;
    const int row  = lane & 15;
    const int hi   = lane >> 4;
    const float* Xb = X + (size_t)bb * xBS;

    v8f acc[4] = {};
    for (int k0 = 0; k0 < K; k0 += 32) {
        __syncthreads();
        // stage A tile (16 m x 32 k): transposed read of adjacency (m contiguous)
#pragma unroll
        for (int it = 0; it < 2; ++it) {
            int i = tid + it * 256;
            int r = i & 15, k = i >> 4;
            int kk = k0 + k;
            float v = (kk < K && (m0 + r) < M) ? Adj[(size_t)kk * ldadj + m0 + r] : 0.0f;
            As[(r << 5) + k] = (__bf16)v;
        }
        // stage X tile into B^T form via float4 loads: Bs[nl][k] = X_b[k0+k, n0+nl]
#pragma unroll
        for (int it = 0; it < 16; ++it) {
            int ch = tid + it * 256;          // 4096 chunks of 4 (along n)
            int k  = ch >> 7;                 // 0..31
            int nl = (ch & 127) << 2;         // 0..508
            int kk = k0 + k;
            int n = n0 + nl, t = n >> 6, c = n & 63;
            float4 v = {};
            if (kk < K) v = *(const float4*)&Xb[(size_t)kk * xVS + t * xTS + c];
            Bs[((nl + 0) << 5) + k] = (__bf16)v.x;
            Bs[((nl + 1) << 5) + k] = (__bf16)v.y;
            Bs[((nl + 2) << 5) + k] = (__bf16)v.z;
            Bs[((nl + 3) << 5) + k] = (__bf16)v.w;
        }
        // prefetch next K slice (adjacency strip + the 8 t-blocks we consume)
        {
            int kpf = k0 + 32 + lane;
            if (kpf < K) {
                __builtin_prefetch(&Adj[(size_t)kpf * ldadj + m0], 0, 0);
                int tpf = (n0 >> 6) + wave;
                __builtin_prefetch(&Xb[(size_t)kpf * xVS + tpf * xTS], 0, 0);
            }
        }
        __syncthreads();

        FragU a;
#pragma unroll
        for (int j = 0; j < 8; ++j) {
            int ka = ((j < 4) ? 0 : 16) + hi * 8 + ((j & 3) << 1);
            a.u[j] = *(const unsigned int*)(As + (row << 5) + ka);
        }
#pragma unroll
        for (int nt = 0; nt < 4; ++nt) {
            const __bf16* Bb = Bs + (wave * 64 + nt * 16) * 32;
            FragU b;
#pragma unroll
            for (int j = 0; j < 8; ++j) {
                int kb = ((j < 4) ? 0 : 16) + hi * 8 + ((j & 3) << 1);
                b.u[j] = *(const unsigned int*)(Bb + (row << 5) + kb);
            }
            acc[nt] = __builtin_amdgcn_wmma_f32_16x16x32_bf16(false, a.v, false, b.v,
                                                              (short)0, acc[nt], false, false);
        }
    }

    float* Ob = Out + (size_t)bb * oBS;
#pragma unroll
    for (int nt = 0; nt < 4; ++nt) {
        const int n = n0 + wave * 64 + nt * 16 + row;
        const int t = n >> 6, c = n & 63;
#pragma unroll
        for (int r = 0; r < 8; ++r) {
            int m = m0 + hi * 8 + r;
            if (m < M) Ob[(size_t)m * oVS + t * oTS + c] = acc[nt][r];
        }
    }
}

// ---------------------------------------------------------------------------
// 3-level Haar DWT along T per (b,n) slice.  T=64 -> no padding needed.
// ---------------------------------------------------------------------------
__global__ __launch_bounds__(256)
void dwt_kernel(const float* __restrict__ x, float* __restrict__ A3w,
                float* __restrict__ D1, float* __restrict__ D2,
                float* __restrict__ D3)
{
    __shared__ float xs[64 * 64];
    __shared__ float lo1[32 * 64];
    __shared__ float lo2[16 * 64];
    const int bn = blockIdx.x;
    const float s = 0.70710678118654752f;
    const float* xp = x + (size_t)bn * 4096;
    for (int i = threadIdx.x; i < 4096; i += 256) xs[i] = xp[i];
    __syncthreads();
    for (int i = threadIdx.x; i < 32 * 64; i += 256) {
        int j = i >> 6, c = i & 63;
        float a = xs[(2 * j) * 64 + c], b = xs[(2 * j + 1) * 64 + c];
        lo1[i] = (a + b) * s;
        D1[(size_t)bn * 2048 + i] = (a - b) * s;
    }
    __syncthreads();
    for (int i = threadIdx.x; i < 16 * 64; i += 256) {
        int j = i >> 6, c = i & 63;
        float a = lo1[(2 * j) * 64 + c], b = lo1[(2 * j + 1) * 64 + c];
        lo2[i] = (a + b) * s;
        D2[(size_t)bn * 1024 + i] = (a - b) * s;
    }
    __syncthreads();
    for (int i = threadIdx.x; i < 8 * 64; i += 256) {
        int j = i >> 6, c = i & 63;
        float a = lo2[(2 * j) * 64 + c], b = lo2[(2 * j + 1) * 64 + c];
        A3w[(size_t)bn * 512 + i] = (a + b) * s;
        D3 [(size_t)bn * 512 + i] = (a - b) * s;
    }
}

// linear interp along time (align_corners=False), Ts -> 64
__device__ __forceinline__ float interp_level(const float* __restrict__ L,
                                              int bn, int Ts, int t, int c)
{
    float pos = ((float)t + 0.5f) * ((float)Ts * (1.0f / 64.0f)) - 0.5f;
    pos = fminf(fmaxf(pos, 0.0f), (float)(Ts - 1));
    int i0 = (int)floorf(pos);
    int i1 = min(i0 + 1, Ts - 1);
    float w = pos - (float)i0;
    float v0 = L[((size_t)bn * Ts + i0) * 64 + c];
    float v1 = L[((size_t)bn * Ts + i1) * 64 + c];
    return v0 * (1.0f - w) + v1 * w;
}

// x_wav = x + sum alpha_i * interp(level_i); written into h[:, 0:64] (ld 320)
__global__ __launch_bounds__(256)
void mcwp_combine(const float* __restrict__ x,
                  const float* __restrict__ A3h, const float* __restrict__ D1h,
                  const float* __restrict__ D2h, const float* __restrict__ D3h,
                  const float* aA3, const float* aD1,
                  const float* aD2, const float* aD3,
                  float* __restrict__ h)
{
    int idx = blockIdx.x * 256 + threadIdx.x;
    if (idx >= 157184 * 64) return;
    int c  = idx & 63;
    int t  = (idx >> 6) & 63;
    int bn = idx >> 12;
    float v = x[idx];
    v += aA3[0] * interp_level(A3h, bn, 8,  t, c);
    v += aD1[0] * interp_level(D1h, bn, 32, t, c);
    v += aD2[0] * interp_level(D2h, bn, 16, t, c);
    v += aD3[0] * interp_level(D3h, bn, 8,  t, c);
    h[((size_t)(bn * 64 + t)) * 320 + c] = v;
}

// diagonal SSM scan along T: S[t] = tanh(a_param[c]) * S[t-1] + u[t]
__global__ __launch_bounds__(64)
void ssm_scan(const float* __restrict__ u, const float* __restrict__ a_param,
              float* __restrict__ S)
{
    const int be = blockIdx.x;       // b*E + e  (1024)
    const int c  = threadIdx.x;      // 64
    const float a = tanhf(a_param[c]);
    float s = 0.0f;
    const float* up = u + (size_t)be * 4096 + c;
    float*       Sp = S + (size_t)be * 4096 + c;
    for (int t = 0; t < 64; ++t) {
        s = a * s + up[t * 64];
        Sp[t * 64] = s;
    }
}

// y_e = gelu(BN(y1)) + xE ; written into h2[:, 0:64] (ld 192)
__global__ __launch_bounds__(256)
void bn_gelu_res(const float* __restrict__ y1, const float* __restrict__ xE,
                 const float* g, const float* bta, const float* rm, const float* rv,
                 float* __restrict__ h2)
{
    int idx = blockIdx.x * 256 + threadIdx.x;
    if (idx >= 4194304) return;
    int c   = idx & 63;
    int row = idx >> 6;
    float y = y1[idx];
    y = (y - rm[c]) * rsqrtf(rv[c] + 1e-5f) * g[c] + bta[c];
    float ge = 0.5f * y * (1.0f + erff(y * 0.70710678118654752f));
    h2[(size_t)row * 192 + c] = ge + xE[idx];
}

// ---------------------------------------------------------------------------
extern "C" void kernel_launch(void* const* d_in, const int* in_sizes, int n_in,
                              void* d_out, int out_size, void* d_ws, size_t ws_size,
                              hipStream_t stream)
{
    (void)in_sizes; (void)n_in; (void)out_size; (void)ws_size;
    const float* x      = (const float*)d_in[0];
    const float* A1     = (const float*)d_in[1];
    const float* A2     = (const float*)d_in[2];
    const float* Hn2e   = (const float*)d_in[3];
    const float* G      = (const float*)d_in[4];
    const float* He2n   = (const float*)d_in[5];
    const float* Wa3    = (const float*)d_in[6];
    const float* ba3    = (const float*)d_in[7];
    const float* Wd1    = (const float*)d_in[8];
    const float* bd1    = (const float*)d_in[9];
    const float* Wd2    = (const float*)d_in[10];
    const float* bd2    = (const float*)d_in[11];
    const float* Wd3    = (const float*)d_in[12];
    const float* bd3    = (const float*)d_in[13];
    const float* aA3    = (const float*)d_in[14];
    const float* aD1    = (const float*)d_in[15];
    const float* aD2    = (const float*)d_in[16];
    const float* aD3    = (const float*)d_in[17];
    const float* Wdgcn  = (const float*)d_in[18];
    const float* bdgcn  = (const float*)d_in[19];
    const float* Wp     = (const float*)d_in[20];
    const float* bp     = (const float*)d_in[21];
    const float* a_par  = (const float*)d_in[22];
    const float* Wb     = (const float*)d_in[23];
    const float* bbp    = (const float*)d_in[24];
    const float* Wg     = (const float*)d_in[25];
    const float* bg     = (const float*)d_in[26];
    const float* Wdd    = (const float*)d_in[27];
    const float* bdd    = (const float*)d_in[28];
    const float* bn_g   = (const float*)d_in[29];
    const float* bn_b   = (const float*)d_in[30];
    const float* bn_rm  = (const float*)d_in[31];
    const float* bn_rv  = (const float*)d_in[32];
    const float* Whg    = (const float*)d_in[33];
    const float* bhg    = (const float*)d_in[34];
    float* out = (float*)d_out;
    float* ws  = (float*)d_ws;

    // workspace layout (floats), with time-based aliasing of region r2
    float* f_h    = ws;                                   // [157184, 320]
    float* f_hout = f_h    + (size_t)157184 * 320;        // [157184, 64]
    float* r2     = f_hout + (size_t)157184 * 64;
    // wavelet stage (aliases r2; dead after mcwp_combine)
    float* f_A3w = r2;
    float* f_D1  = f_A3w + (size_t)19648 * 64;
    float* f_D2  = f_D1  + (size_t)78592 * 64;
    float* f_D3  = f_D2  + (size_t)39296 * 64;
    float* f_A3h = f_D3  + (size_t)19648 * 64;
    float* f_D1h = f_A3h + (size_t)19648 * 64;
    float* f_D2h = f_D1h + (size_t)78592 * 64;
    float* f_D3h = f_D2h + (size_t)39296 * 64;
    // edge stage (aliases r2; starts after f_hout is produced)
    float* f_xe = r2;                                     // [65536, 64]
    float* f_xE = f_xe + (size_t)65536 * 64;
    float* f_u  = f_xE + (size_t)65536 * 64;
    float* f_S  = f_u  + (size_t)65536 * 64;
    float* f_y1 = f_S  + (size_t)65536 * 64;
    float* f_h2 = f_y1 + (size_t)65536 * 64;              // [65536, 192]
    float* f_he = f_h2 + (size_t)65536 * 192;             // [65536, 64]

    // ---- MCWP -------------------------------------------------------------
    dwt_kernel<<<2456, 256, 0, stream>>>(x, f_A3w, f_D1, f_D2, f_D3);
    conv_gemm_wmma<64, false><<<154, 256, 0, stream>>>(f_A3w, 64, Wa3, ba3, f_A3h, 19648);
    conv_gemm_wmma<64, false><<<614, 256, 0, stream>>>(f_D1,  64, Wd1, bd1, f_D1h, 78592);
    conv_gemm_wmma<64, false><<<307, 256, 0, stream>>>(f_D2,  64, Wd2, bd2, f_D2h, 39296);
    conv_gemm_wmma<64, false><<<154, 256, 0, stream>>>(f_D3,  64, Wd3, bd3, f_D3h, 19648);
    mcwp_combine<<<39296, 256, 0, stream>>>(x, f_A3h, f_D1h, f_D2h, f_D3h,
                                            aA3, aD1, aD2, aD3, f_h);

    // ---- DGCN: diffusions write directly into the 320-wide concat --------
    dim3 gdN(20, 8, 8);    // ceil(307/16) x (4096/512) x B
    diffuse_gemm_wmma<307><<<gdN, 256, 0, stream>>>(A1, 307, 307,
        f_h + 0,   6287360, 20480, 320,  f_h + 64,  6287360, 20480, 320);
    diffuse_gemm_wmma<307><<<gdN, 256, 0, stream>>>(A1, 307, 307,
        f_h + 64,  6287360, 20480, 320,  f_h + 128, 6287360, 20480, 320);
    diffuse_gemm_wmma<307><<<gdN, 256, 0, stream>>>(A2, 307, 307,
        f_h + 0,   6287360, 20480, 320,  f_h + 192, 6287360, 20480, 320);
    diffuse_gemm_wmma<307><<<gdN, 256, 0, stream>>>(A2, 307, 307,
        f_h + 192, 6287360, 20480, 320,  f_h + 256, 6287360, 20480, 320);
    conv_gemm_wmma<320, false><<<1228, 256, 0, stream>>>(f_h, 320, Wdgcn, bdgcn, f_hout, 157184);

    // ---- node -> edge -----------------------------------------------------
    dim3 gdE(8, 8, 8);     // ceil(128/16) x (4096/512) x B
    diffuse_gemm_wmma<307><<<gdE, 256, 0, stream>>>(Hn2e, 128, 128,
        f_hout, 1257472, 4096, 64,  f_xe, 524288, 4096, 64);

    // ---- EdgeTemporalSSM --------------------------------------------------
    conv_gemm_wmma<64, false><<<512, 256, 0, stream>>>(f_xe, 64, Wp, bp,  f_xE, 65536);
    conv_gemm_wmma<64, false><<<512, 256, 0, stream>>>(f_xE, 64, Wb, bbp, f_u,  65536);
    ssm_scan<<<1024, 64, 0, stream>>>(f_u, a_par, f_S);
    conv_gemm_wmma<64, false><<<512, 256, 0, stream>>>(f_S,  64, Wg, bg,  f_y1, 65536);
    conv_gemm_wmma<64, true ><<<512, 256, 0, stream>>>(f_xE, 64, Wdd, bdd, f_y1, 65536); // accumulate
    bn_gelu_res<<<16384, 256, 0, stream>>>(f_y1, f_xE, bn_g, bn_b, bn_rm, bn_rv, f_h2);

    // ---- HGCN: diffusions into 192-wide concat ---------------------------
    diffuse_gemm_wmma<128><<<gdE, 256, 0, stream>>>(G, 128, 128,
        f_h2 + 0,  1572864, 12288, 192,  f_h2 + 64,  1572864, 12288, 192);
    diffuse_gemm_wmma<128><<<gdE, 256, 0, stream>>>(G, 128, 128,
        f_h2 + 64, 1572864, 12288, 192,  f_h2 + 128, 1572864, 12288, 192);
    conv_gemm_wmma<192, false><<<512, 256, 0, stream>>>(f_h2, 192, Whg, bhg, f_he, 65536);

    // ---- edge -> node (final output) -------------------------------------
    dim3 gdO(20, 8, 8);
    diffuse_gemm_wmma<128><<<gdO, 256, 0, stream>>>(He2n, 307, 307,
        f_he, 524288, 4096, 64,  out, 1257472, 4096, 64);
}
